// Attention_6992206758310
// MI455X (gfx1250) — matrix operand order repaired and checked by reading, not compiled, not run on hardware
//
#include <hip/hip_runtime.h>
#include <hip/hip_bf16.h>

// ---------------------------------------------------------------------------
// MI455X (gfx1250) attention block: LN -> QKV GEMM -> RoPE -> flash attn -> out proj
// Matmuls on v_wmma_f32_16x16x32_bf16 (wave32); GEMM tiles double-buffered in
// LDS via global_load_async_to_lds_b128 (ASYNCcnt, in-order -> partial waits).
// ---------------------------------------------------------------------------

typedef __attribute__((ext_vector_type(16))) __bf16 v16bf;
typedef __attribute__((ext_vector_type(8)))  float  v8f;

#define NN   2048
#define DIM  1024
#define NH   16
#define DH   64

union Frag {
    v16bf v;
    unsigned int   u[8];
    unsigned short h[16];
    uint4          q[2];
};

static __device__ inline v8f v8f_zero() {
    v8f z = {0.f,0.f,0.f,0.f,0.f,0.f,0.f,0.f};
    return z;
}

static __device__ inline unsigned short f2bf(float f) {
    unsigned int u = __float_as_uint(f);
    u += 0x7FFFu + ((u >> 16) & 1u);     // round-to-nearest-even
    return (unsigned short)(u >> 16);
}
static __device__ inline unsigned int pack2bf(float a, float b) {
    return (unsigned int)f2bf(a) | ((unsigned int)f2bf(b) << 16);
}

static __device__ inline v8f wmma_bf16(const Frag& a, const Frag& b, v8f c) {
    return __builtin_amdgcn_wmma_f32_16x16x32_bf16(
        false, a.v, false, b.v, (short)0, c, false, false);
}

// Async copy 32 bytes global->LDS (two b128 transfers, ASYNCcnt).
// IOFFSET is added to BOTH the LDS and global addresses (ISA 10.4 pseudocode).
static __device__ inline void async_copy_b256(unsigned lds_addr, const void* gptr) {
    unsigned long long ga = (unsigned long long)gptr;
    asm volatile("global_load_async_to_lds_b128 %0, %1, off"
                 :: "v"(lds_addr), "v"(ga) : "memory");
    asm volatile("global_load_async_to_lds_b128 %0, %1, off offset:16"
                 :: "v"(lds_addr), "v"(ga) : "memory");
}
static __device__ inline void async_wait0() {
    asm volatile("s_wait_asynccnt 0" ::: "memory");
}
// Async loads complete in-order per wave: <=4 outstanding means the 4 oldest
// (previous tile) have landed while the 4 newest (next tile) stay in flight.
static __device__ inline void async_wait4() {
    asm volatile("s_wait_asynccnt 4" ::: "memory");
}
static __device__ inline unsigned lds_off32(const void* p) {
    // flat shared address: low 32 bits are the LDS byte offset
    return (unsigned)(size_t)p;
}

// ---------------------------------------------------------------------------
// K1: LayerNorm, f32 -> bf16.  One block per row, 256 threads, 4 elems each.
// ---------------------------------------------------------------------------
__global__ __launch_bounds__(256) void k_layernorm(
    const float* __restrict__ x, const float* __restrict__ gamma,
    const float* __restrict__ beta, unsigned short* __restrict__ xn)
{
    const int row = blockIdx.x;
    const int t   = threadIdx.x;
    const float4 v = ((const float4*)(x + (size_t)row * DIM))[t];

    __shared__ float red[8];
    __shared__ float bc;

    float s = v.x + v.y + v.z + v.w;
    for (int off = 16; off > 0; off >>= 1) s += __shfl_down(s, off);
    if ((t & 31) == 0) red[t >> 5] = s;
    __syncthreads();
    if (t == 0) { float a = 0.f; for (int i = 0; i < 8; ++i) a += red[i]; bc = a; }
    __syncthreads();
    const float mean = bc * (1.f / DIM);

    float d0 = v.x - mean, d1 = v.y - mean, d2 = v.z - mean, d3 = v.w - mean;
    float s2 = d0*d0 + d1*d1 + d2*d2 + d3*d3;
    for (int off = 16; off > 0; off >>= 1) s2 += __shfl_down(s2, off);
    __syncthreads();
    if ((t & 31) == 0) red[t >> 5] = s2;
    __syncthreads();
    if (t == 0) { float a = 0.f; for (int i = 0; i < 8; ++i) a += red[i]; bc = a; }
    __syncthreads();
    const float rstd = rsqrtf(bc * (1.f / DIM) + 1e-6f);

    const int c = t * 4;
    float y0 = d0 * rstd * gamma[c+0] + beta[c+0];
    float y1 = d1 * rstd * gamma[c+1] + beta[c+1];
    float y2 = d2 * rstd * gamma[c+2] + beta[c+2];
    float y3 = d3 * rstd * gamma[c+3] + beta[c+3];
    uint2 p; p.x = pack2bf(y0, y1); p.y = pack2bf(y2, y3);
    ((uint2*)(xn + (size_t)row * DIM))[t] = p;
}

// ---------------------------------------------------------------------------
// K2: f32 [K=1024][N] -> bf16 transposed [N][K=1024].  One-time; writes
// coalesced (k fastest), strided reads stay in L2.
// ---------------------------------------------------------------------------
__global__ void k_f2bf_t(const float* __restrict__ in,
                         unsigned short* __restrict__ out, int N)
{
    int o = blockIdx.x * blockDim.x + threadIdx.x;   // output index: n*1024 + k
    if (o < N * 1024) {
        int n = o >> 10, k = o & 1023;
        out[o] = f2bf(in[(size_t)k * N + n]);
    }
}

// ---------------------------------------------------------------------------
// K3/K6: C[M,N] = A[M,K] * Bt[N,K]^T (+bias), bf16 in, f32 out.
// Block tile 128x128, 8 waves in 2x4 (M x N), wave tile 64x32, K-step 32.
// Double-buffered LDS tiles filled by async-to-LDS; s_wait_asynccnt 4 lets
// the next tile's fill run behind the current tile's WMMAs.
// ---------------------------------------------------------------------------
#define BM 128
#define BN 128
#define BK 32

__global__ __launch_bounds__(256) void k_gemm_bf16(
    const unsigned short* __restrict__ A, const unsigned short* __restrict__ Bt,
    float* __restrict__ C, const float* __restrict__ bias,
    int M, int N, int K)
{
    alignas(16) __shared__ unsigned short Al[2][BM][BK];
    alignas(16) __shared__ unsigned short Bl[2][BN][BK];

    const int bn = blockIdx.x * BN;
    const int bm = blockIdx.y * BM;
    const int t    = threadIdx.x;
    const int wave = t >> 5;
    const int lane = t & 31;
    const int wm   = wave >> 2;        // 0..1
    const int wn   = wave & 3;         // 0..3
    const int sub  = lane >> 4;        // half-wave
    const int l15  = lane & 15;

    v8f acc[4][2];
    #pragma unroll
    for (int i = 0; i < 4; ++i)
        #pragma unroll
        for (int j = 0; j < 2; ++j) acc[i][j] = v8f_zero();

    // cooperative staging: thread -> (row, 16-elem k-segment)
    const int srow = t >> 1, sseg = (t & 1) * 16;
    const unsigned lds_a0 = lds_off32(&Al[0][srow][sseg]);
    const unsigned lds_a1 = lds_off32(&Al[1][srow][sseg]);
    const unsigned lds_b0 = lds_off32(&Bl[0][srow][sseg]);
    const unsigned lds_b1 = lds_off32(&Bl[1][srow][sseg]);
    const unsigned short* gA = A  + (size_t)(bm + srow) * K + sseg;
    const unsigned short* gB = Bt + (size_t)(bn + srow) * K + sseg;

    // prologue: fill buffer 0
    async_copy_b256(lds_a0, gA);
    async_copy_b256(lds_b0, gB);

    for (int k0 = 0; k0 < K; k0 += BK) {
        const int cur = (k0 / BK) & 1;
        if (k0 + BK < K) {
            // issue next tile into the other buffer, keep it in flight
            async_copy_b256(cur ? lds_a0 : lds_a1, gA + k0 + BK);
            async_copy_b256(cur ? lds_b0 : lds_b1, gB + k0 + BK);
            async_wait4();          // current tile landed; next tile async
        } else {
            async_wait0();
        }
        __syncthreads();            // all waves' current tile visible

        const unsigned short (*Ac)[BK] = Al[cur];
        const unsigned short (*Bc)[BK] = Bl[cur];

        // ---- fragments + WMMA ----
        Frag af[4], bf_[2];
        const int kb = 8 * sub;
        #pragma unroll
        for (int mt = 0; mt < 4; ++mt) {
            const int r = wm * 64 + mt * 16 + l15;
            af[mt].q[0] = *(const uint4*)&Ac[r][kb];
            af[mt].q[1] = *(const uint4*)&Ac[r][16 + kb];
        }
        #pragma unroll
        for (int nt = 0; nt < 2; ++nt) {
            const int c = wn * 32 + nt * 16 + l15;
            bf_[nt].q[0] = *(const uint4*)&Bc[c][kb];
            bf_[nt].q[1] = *(const uint4*)&Bc[c][16 + kb];
        }
        #pragma unroll
        for (int mt = 0; mt < 4; ++mt)
            #pragma unroll
            for (int nt = 0; nt < 2; ++nt)
                acc[mt][nt] = wmma_bf16(af[mt], bf_[nt], acc[mt][nt]);

        __syncthreads();            // buffer 'cur' free for tile k+2's fill
    }

    // ---- epilogue: C layout row = reg + 8*sub, col = l15 ----
    #pragma unroll
    for (int mt = 0; mt < 4; ++mt)
        #pragma unroll
        for (int nt = 0; nt < 2; ++nt) {
            const int col = bn + wn * 32 + nt * 16 + l15;
            const float bv = bias ? bias[col] : 0.f;
            #pragma unroll
            for (int r = 0; r < 8; ++r) {
                const int row = bm + wm * 64 + mt * 16 + 8 * sub + r;
                C[(size_t)row * N + col] = acc[mt][nt][r] + bv;
            }
        }
}

// ---------------------------------------------------------------------------
// K4: RoPE on q,k rows 1..N-1; emit q,k row-major bf16 and V transposed
//     vt[h*64+d][n] bf16 (so attention PV B-fragments are contiguous).
// 512 threads/row: thread t -> head t>>5, pair t&31.
// ---------------------------------------------------------------------------
__global__ __launch_bounds__(512) void k_rope(
    const float* __restrict__ qkv, const float* __restrict__ psin,
    const float* __restrict__ pcos, unsigned short* __restrict__ qb,
    unsigned short* __restrict__ kb, unsigned short* __restrict__ vt)
{
    const int n = blockIdx.x;
    const int t = threadIdx.x;
    const int p = t & 31;
    const int col = 2 * t;                      // == h*64 + 2*p
    const float* row = qkv + (size_t)n * (3 * DIM);

    float q0 = row[col],          q1 = row[col + 1];
    float k0 = row[DIM + col],    k1 = row[DIM + col + 1];
    float v0 = row[2*DIM + col],  v1 = row[2*DIM + col + 1];

    if (n > 0) {
        const float s = psin[(size_t)(n - 1) * 32 + p];
        const float c = pcos[(size_t)(n - 1) * 32 + p];
        float tq0 = q0 * c - q1 * s, tq1 = q1 * c + q0 * s;
        float tk0 = k0 * c - k1 * s, tk1 = k1 * c + k0 * s;
        q0 = tq0; q1 = tq1; k0 = tk0; k1 = tk1;
    }
    ((unsigned int*)qb)[(size_t)n * (DIM/2) + t] = pack2bf(q0, q1);
    ((unsigned int*)kb)[(size_t)n * (DIM/2) + t] = pack2bf(k0, k1);
    vt[(size_t)col * NN + n]       = f2bf(v0);
    vt[(size_t)(col + 1) * NN + n] = f2bf(v1);
}

// ---------------------------------------------------------------------------
// K5: flash attention. One wave = (16-query tile, head). Key tiles of 32.
// S tile via 4 WMMAs; online softmax with 16-lane xor reductions;
// P goes through a per-wave LDS tile to switch C-layout -> A-layout;
// O = 4 WMMA accumulators (16x64).
// Mask: allowed(i,j) = ((j<=i) || i==0 || j==0) && pad[j].
// ---------------------------------------------------------------------------
__global__ __launch_bounds__(128) void k_attn(
    const unsigned short* __restrict__ Q, const unsigned short* __restrict__ Km,
    const unsigned short* __restrict__ Vt, const int* __restrict__ mask,
    unsigned short* __restrict__ O)
{
    alignas(16) __shared__ unsigned short Pl[4][16][32];

    const int wave = threadIdx.x >> 5;
    const int lane = threadIdx.x & 31;
    const int sub  = lane >> 4;
    const int l15  = lane & 15;
    const int h    = blockIdx.x;
    const int qt   = blockIdx.y * 4 + wave;
    const int qi0  = qt * 16;
    const float scale = 0.03125f;               // 1024^-0.5

    // Q fragments (16x64 = 2 x 16x32 A fragments), rows qi0..qi0+15.
    // A-layout per lane: k in [kb, kb+8) and [16+kb, 16+kb+8), kb = 8*sub —
    // two contiguous 16B chunks.
    Frag qa[2];
    {
        const unsigned short* qp = Q + (size_t)(qi0 + l15) * DIM + h * DH;
        const int kb = 8 * sub;
        #pragma unroll
        for (int f = 0; f < 2; ++f) {
            qa[f].q[0] = *(const uint4*)(qp + f * 32 + kb);
            qa[f].q[1] = *(const uint4*)(qp + f * 32 + 16 + kb);
        }
    }

    v8f o[4]; float m_r[8], l_r[8];
    #pragma unroll
    for (int c = 0; c < 4; ++c) o[c] = v8f_zero();
    #pragma unroll
    for (int r = 0; r < 8; ++r) { m_r[r] = -INFINITY; l_r[r] = 0.f; }

    const int nkt = (qi0 == 0) ? (NN / 32) : ((qi0 + 15) / 32 + 1);

    for (int kt = 0; kt < nkt; ++kt) {
        const int j0 = kt * 32;

        // ---- S = Q * K^T (two 16x16 tiles over 32 keys) ----
        v8f s[2]; s[0] = v8f_zero(); s[1] = v8f_zero();
        #pragma unroll
        for (int c = 0; c < 2; ++c) {
            const int j = j0 + 16 * c + l15;
            #pragma unroll
            for (int f = 0; f < 2; ++f) {
                Frag bf_;   // B-layout: lane = col j, k in [8*sub,+8) and [16+8*sub,+8)
                const unsigned short* kp =
                    Km + (size_t)j * DIM + h * DH + f * 32 + 8 * sub;
                bf_.q[0] = *(const uint4*)(kp);
                bf_.q[1] = *(const uint4*)(kp + 16);
                s[c] = wmma_bf16(qa[f], bf_, s[c]);
            }
        }

        // ---- scale + mask, per-row tile max ----
        float mt_[8];
        #pragma unroll
        for (int r = 0; r < 8; ++r) mt_[r] = -INFINITY;
        #pragma unroll
        for (int c = 0; c < 2; ++c) {
            const int j = j0 + 16 * c + l15;
            const bool padj = (j == 0) ? true : (mask[j - 1] != 0);
            #pragma unroll
            for (int r = 0; r < 8; ++r) {
                const int i = qi0 + 8 * sub + r;
                float val = s[c][r] * scale;
                const bool allowed = padj && ((j <= i) || (i == 0) || (j == 0));
                val = allowed ? val : -1.0e10f;
                s[c][r] = val;
                mt_[r] = fmaxf(mt_[r], val);
            }
        }
        #pragma unroll
        for (int r = 0; r < 8; ++r) {
            float v = mt_[r];
            for (int off = 1; off < 16; off <<= 1) v = fmaxf(v, __shfl_xor(v, off));
            mt_[r] = v;
        }

        // ---- online softmax update ----
        float alpha[8];
        #pragma unroll
        for (int r = 0; r < 8; ++r) {
            const float mn = fmaxf(m_r[r], mt_[r]);
            alpha[r] = __expf(m_r[r] - mn);
            m_r[r] = mn;
        }
        float rs[8];
        #pragma unroll
        for (int r = 0; r < 8; ++r) rs[r] = 0.f;
        #pragma unroll
        for (int c = 0; c < 2; ++c)
            #pragma unroll
            for (int r = 0; r < 8; ++r) {
                const float p = __expf(s[c][r] - m_r[r]);
                rs[r] += p;
                Pl[wave][8 * sub + r][16 * c + l15] = f2bf(p);
            }
        #pragma unroll
        for (int r = 0; r < 8; ++r) {
            float v = rs[r];
            for (int off = 1; off < 16; off <<= 1) v += __shfl_xor(v, off);
            l_r[r] = l_r[r] * alpha[r] + v;
        }
        #pragma unroll
        for (int c = 0; c < 4; ++c)
            #pragma unroll
            for (int r = 0; r < 8; ++r) o[c][r] *= alpha[r];

        // intra-wave LDS ordering (DS ops are in-order per wave; pins IR order)
        asm volatile("s_wait_dscnt 0" ::: "memory");

        // ---- P (A-layout from LDS), O += P * V ----
        Frag pa;
        {
            const int kb = 8 * sub;
            pa.q[0] = *(const uint4*)&Pl[wave][l15][kb];
            pa.q[1] = *(const uint4*)&Pl[wave][l15][16 + kb];
        }
        #pragma unroll
        for (int c2 = 0; c2 < 4; ++c2) {
            Frag vb;    // B-layout: lane = col d, keys [8*sub,+8) and [16+8*sub,+8) in Vt[h,d][*]
            const int d = 16 * c2 + l15;
            const unsigned short* vp = Vt + (size_t)(h * DH + d) * NN + j0 + 8 * sub;
            vb.q[0] = *(const uint4*)(vp);
            vb.q[1] = *(const uint4*)(vp + 16);
            o[c2] = wmma_bf16(pa, vb, o[c2]);
        }
    }

    // ---- normalize + store bf16, row-major [n][h*64+d] ----
    #pragma unroll
    for (int c2 = 0; c2 < 4; ++c2) {
        const int d = 16 * c2 + l15;
        #pragma unroll
        for (int r = 0; r < 8; ++r) {
            const int i = qi0 + 8 * sub + r;
            O[(size_t)i * DIM + h * DH + d] = f2bf(o[c2][r] / l_r[r]);
        }
    }
}

// ---------------------------------------------------------------------------
// Launch
// ---------------------------------------------------------------------------
extern "C" void kernel_launch(void* const* d_in, const int* in_sizes, int n_in,
                              void* d_out, int out_size, void* d_ws, size_t ws_size,
                              hipStream_t stream)
{
    const float* x        = (const float*)d_in[0];
    const float* pos_sin  = (const float*)d_in[1];
    const float* pos_cos  = (const float*)d_in[2];
    const int*   mask     = (const int*)  d_in[3];
    const float* ln_scale = (const float*)d_in[4];
    const float* ln_bias  = (const float*)d_in[5];
    const float* w_qkv    = (const float*)d_in[6];
    const float* w_out    = (const float*)d_in[7];
    const float* b_out    = (const float*)d_in[8];
    float* out = (float*)d_out;

    // workspace carve-up (~54.6 MB)
    char* ws = (char*)d_ws;
    size_t off = 0;
    unsigned short* xn     = (unsigned short*)(ws + off); off += (size_t)NN * DIM * 2;
    unsigned short* wqkv_t = (unsigned short*)(ws + off); off += (size_t)DIM * 3 * DIM * 2;
    unsigned short* wout_t = (unsigned short*)(ws + off); off += (size_t)DIM * DIM * 2;
    float*          qkv    = (float*)         (ws + off); off += (size_t)NN * 3 * DIM * 4;
    unsigned short* qb     = (unsigned short*)(ws + off); off += (size_t)NN * DIM * 2;
    unsigned short* kb     = (unsigned short*)(ws + off); off += (size_t)NN * DIM * 2;
    unsigned short* vt     = (unsigned short*)(ws + off); off += (size_t)NN * DIM * 2;
    unsigned short* attn   = (unsigned short*)(ws + off); off += (size_t)NN * DIM * 2;

    // 1) LayerNorm -> bf16
    k_layernorm<<<NN, 256, 0, stream>>>(x, ln_scale, ln_bias, xn);

    // 2) weights -> bf16, transposed to [N][K]
    {
        int nq = 3 * DIM * DIM;
        k_f2bf_t<<<(nq + 255) / 256, 256, 0, stream>>>(w_qkv, wqkv_t, 3 * DIM);
        int no = DIM * DIM;
        k_f2bf_t<<<(no + 255) / 256, 256, 0, stream>>>(w_out, wout_t, DIM);
    }

    // 3) qkv = xn @ w_qkv   (2048 x 3072 x 1024)
    k_gemm_bf16<<<dim3(3 * DIM / BN, NN / BM), 256, 0, stream>>>(
        xn, wqkv_t, qkv, nullptr, NN, 3 * DIM, DIM);

    // 4) RoPE + layout (q,k row-major bf16; V transposed [h*64+d][n])
    k_rope<<<NN, 512, 0, stream>>>(qkv, pos_sin, pos_cos, qb, kb, vt);

    // 5) flash attention -> attn bf16 [n][h*64+d]
    k_attn<<<dim3(NH, NN / 16 / 4), 128, 0, stream>>>(qb, kb, vt, mask, attn);

    // 6) out = attn @ w_out + b_out   (2048 x 1024 x 1024), f32
    k_gemm_bf16<<<dim3(DIM / BN, NN / BM), 256, 0, stream>>>(
        attn, wout_t, out, b_out, NN, DIM, DIM);
}
